// EuclideanAttention_59485297050331
// MI455X (gfx1250) — matrix operand order repaired and checked by reading, not compiled, not run on hardware
//
#include <hip/hip_runtime.h>

// ---------------------------------------------------------------------------
// EuclideanAttention on MI455X (gfx1250, wave32, WMMA bf16 -> f32)
//   B=2, S=2048, E=1024, H=16, D=64
// All matmul FLOPs through v_wmma_f32_16x16x32_bf16; every fragment load is
// per-lane contiguous (lowers to b128); weights and V pre-transposed; K-loops
// software-pipelined so fragment loads overlap WMMA / softmax work.
// ---------------------------------------------------------------------------

#define BATCH     2
#define SEQ       2048
#define EMBED     1024
#define HEADS     16
#define HDIM      64
#define MTOT      (BATCH * SEQ)      // 4096 rows for token-major GEMMs

typedef __attribute__((ext_vector_type(16))) __bf16 v16bf;
typedef __attribute__((ext_vector_type(8)))  float  v8f;

// ----------------------------- bf16 helpers --------------------------------
__device__ __forceinline__ unsigned short f2bf_bits(float f) {
  unsigned int u = __builtin_bit_cast(unsigned int, f);
  u += 0x7fffu + ((u >> 16) & 1u);            // round-to-nearest-even
  return (unsigned short)(u >> 16);
}
__device__ __forceinline__ float bfbits2f(unsigned int hs) {
  unsigned int u = hs << 16;
  return __builtin_bit_cast(float, u);
}
__device__ __forceinline__ unsigned int pack2bf(float a, float b) {
  return (unsigned int)f2bf_bits(a) | ((unsigned int)f2bf_bits(b) << 16);
}

// ----------------------------- WMMA wrapper --------------------------------
__device__ __forceinline__ v8f wmma_bf16(v16bf a, v16bf b, v8f c) {
  // (neg_a, A, neg_b, B, c_mod, C, reuse_a, reuse_b)
  return __builtin_amdgcn_wmma_f32_16x16x32_bf16(false, a, false, b,
                                                 (short)0, c, false, false);
}

// A fragment: 16x32 (MxK), row-major source, row stride lda (bf16 elems).
// Per lane (m = lane&15, half = lane>>4) the 16 elements are two contiguous
// 8-element (16B) runs: k = half*8 + {0..7} and k = 16 + half*8 + {0..7}.
__device__ __forceinline__ v16bf load_a_frag(const unsigned short* a, int lda) {
  const int lane = threadIdx.x & 31;
  const int m = lane & 15, half = lane >> 4;
  v16bf f;
#pragma unroll
  for (int p = 0; p < 8; ++p) {
    const int k = (p < 4) ? (half * 8 + 2 * p) : (16 + half * 8 + 2 * (p - 4));
    f[2 * p]     = __builtin_bit_cast(__bf16, a[m * lda + k]);
    f[2 * p + 1] = __builtin_bit_cast(__bf16, a[m * lda + k + 1]);
  }
  return f;
}

// B fragment: 32x16 (KxN) where source stores element(k,n) at mem[n*ldb + k]
// (i.e. N-major / "transposed" storage). Per lane (n = lane&15, half) the 16
// elements are two contiguous 8-element (16B) runs: k = half*8 + {0..7} and k = 16 + half*8 + {0..7}.
__device__ __forceinline__ v16bf load_bt_frag(const unsigned short* b, int ldb) {
  const int lane = threadIdx.x & 31;
  const int n = lane & 15, half = lane >> 4;
  v16bf f;
#pragma unroll
  for (int v = 0; v < 8; ++v) {
    const int k = (v < 4) ? (half * 8 + 2 * v) : (16 + half * 8 + 2 * (v - 4));
    f[2 * v]     = __builtin_bit_cast(__bf16, b[n * ldb + k]);
    f[2 * v + 1] = __builtin_bit_cast(__bf16, b[n * ldb + k + 1]);
  }
  return f;
}

// --------------------------- fp32 -> bf16 convert --------------------------
__global__ __launch_bounds__(256) void cvt_bf16_kernel(
    const float* __restrict__ in, unsigned short* __restrict__ out, int n) {
  int i = blockIdx.x * 256 + threadIdx.x;
  if (i < n) out[i] = f2bf_bits(in[i]);
}

// ------------------- fp32 [K,N] -> bf16 [N,K] transpose --------------------
__global__ __launch_bounds__(256) void transpose_cvt_kernel(
    const float* __restrict__ in, unsigned short* __restrict__ out) {
  __shared__ unsigned short t[32][33];
  const int tx = threadIdx.x & 31;
  const int ty = threadIdx.x >> 5;              // 0..7
  const int n0 = blockIdx.x * 32;
  const int k0 = blockIdx.y * 32;
#pragma unroll
  for (int i = 0; i < 4; ++i) {
    const int k = ty * 4 + i;
    t[tx][k] = f2bf_bits(in[(k0 + k) * EMBED + n0 + tx]);
  }
  __syncthreads();
#pragma unroll
  for (int i = 0; i < 4; ++i) {
    const int n = ty * 4 + i;
    out[(n0 + n) * EMBED + k0 + tx] = t[n][tx];
  }
}

// --------------------------- row sum of squares ----------------------------
// X: head-major bf16 [B*H*S, 64]; out: fp32 [B*H*S]
__global__ __launch_bounds__(256) void rowsq_kernel(
    const unsigned short* __restrict__ X, float* __restrict__ out) {
  int row = blockIdx.x * 256 + threadIdx.x;
  const unsigned int* p = (const unsigned int*)(X + row * HDIM);
  float s = 0.f;
#pragma unroll
  for (int i = 0; i < HDIM / 2; ++i) {
    unsigned int u = p[i];
    float a = bfbits2f(u & 0xffffu);
    float b = bfbits2f(u >> 16);
    s += a * a + b * b;
  }
  out[row] = s;
}

// ------------------------------- GEMM --------------------------------------
// C[M=4096, N=1024] = A_bf16[M,K] @ Wt_bf16[N,K]^T + bias.
// LDS-free, software-pipelined: two fragment stages in flight so WMMAs of one
// stage overlap the b128 loads of the other; L2 (192 MB) carries all reuse.
// mode 0: fp32 token-major out (final projection -> d_out)
// mode 1: bf16 head-major [B,H,S,D]   (Q/K projections)
// mode 2: bf16 head-transposed [B,H,D,S] (V projection; packed dword stores)
__global__ __launch_bounds__(256) void gemm_bf16_kernel(
    const unsigned short* __restrict__ A,    // [4096,1024] bf16 row-major
    const unsigned short* __restrict__ Wt,   // [1024,1024] bf16, [N,K]
    const float* __restrict__ bias,          // [1024] fp32
    float* __restrict__ outF,                // mode 0
    unsigned short* __restrict__ outH,       // mode 1 / 2
    int mode) {
  const int tid  = threadIdx.x;
  const int wave = tid >> 5, lane = tid & 31;
  const int half = lane >> 4, nl = lane & 15;
  const int n0 = blockIdx.x * 64;
  const int m0 = blockIdx.y * 128;
  const int mrow = m0 + wave * 16;

  v8f acc[4] = {};

  const unsigned short* Arow = A + mrow * EMBED;
  const unsigned short* Wrow = Wt + n0 * EMBED;

  // stage-0 fragments for k0 = 0
  v16bf af0 = load_a_frag(Arow, EMBED);
  v16bf bf0[4];
#pragma unroll
  for (int t = 0; t < 4; ++t)
    bf0[t] = load_bt_frag(Wrow + (t * 16) * EMBED, EMBED);

  for (int k0 = 0; k0 < EMBED; k0 += 64) {
    // stage-1 loads (k0+32) in flight over stage-0 WMMAs
    v16bf af1 = load_a_frag(Arow + k0 + 32, EMBED);
    v16bf bf1[4];
#pragma unroll
    for (int t = 0; t < 4; ++t)
      bf1[t] = load_bt_frag(Wrow + (t * 16) * EMBED + k0 + 32, EMBED);

    if (k0 + 128 < EMBED) {  // keep HBM->L2 two double-stages ahead
      __builtin_prefetch(Arow + nl * EMBED + k0 + 128, 0, 3);
      __builtin_prefetch(Wrow + nl * EMBED + k0 + 128, 0, 3);
    }

#pragma unroll
    for (int t = 0; t < 4; ++t) acc[t] = wmma_bf16(af0, bf0[t], acc[t]);

    // stage-0 loads for next double-step (k0+64) over stage-1 WMMAs
    if (k0 + 64 < EMBED) {
      af0 = load_a_frag(Arow + k0 + 64, EMBED);
#pragma unroll
      for (int t = 0; t < 4; ++t)
        bf0[t] = load_bt_frag(Wrow + (t * 16) * EMBED + k0 + 64, EMBED);
    }

#pragma unroll
    for (int t = 0; t < 4; ++t) acc[t] = wmma_bf16(af1, bf1[t], acc[t]);
  }

  // epilogue (C layout: m = half*8 + r, n = nl within each 16x16 tile)
  const int bidx = m0 >> 11;                       // batch (m0 mult of 128)
  const int srow = (mrow + half * 8) & (SEQ - 1);  // first seq row of this lane
#pragma unroll
  for (int t = 0; t < 4; ++t) {
    const int n = n0 + t * 16 + nl;
    const float bv = bias[n];
    if (mode == 2) {
      // Vt[b][h][d][s]: consecutive r -> consecutive s -> packed dword stores
      const int h = n >> 6, d = n & (HDIM - 1);
      unsigned int* o32 = (unsigned int*)(outH +
          (((size_t)(bidx * HEADS + h) * HDIM + d) * SEQ + srow));
#pragma unroll
      for (int j = 0; j < 4; ++j)
        o32[j] = pack2bf(acc[t][2 * j] + bv, acc[t][2 * j + 1] + bv);
    } else {
#pragma unroll
      for (int r = 0; r < 8; ++r) {
        const int m = mrow + half * 8 + r;
        const float v = acc[t][r] + bv;
        if (mode == 0) {
          outF[m * EMBED + n] = v;
        } else {
          const int s = m & (SEQ - 1);
          const int h = n >> 6, d = n & (HDIM - 1);
          outH[(((bidx * HEADS + h) * SEQ) + s) * HDIM + d] = f2bf_bits(v);
        }
      }
    }
  }
}

// --------------------------- flash attention -------------------------------
// One block per (b*H+h, 128 query rows). 8 waves, 16 query rows each.
// scores = -max(|q|^2 + |k|^2 - 2 qk, 0)/T ; online softmax ; O = P @ V.
// Pipeline: V frags for tile j issued first (consumed after softmax); K frags
// for tile j+1 issued right after the score WMMAs (overlap softmax VALU).
__global__ __launch_bounds__(256) void attn_kernel(
    const unsigned short* __restrict__ Qh,   // [B,H,S,D] bf16
    const unsigned short* __restrict__ Kh,   // [B,H,S,D] bf16
    const unsigned short* __restrict__ Vt,   // [B,H,D,S] bf16
    const float* __restrict__ qsq,           // [B*H*S]
    const float* __restrict__ ksq,           // [B*H*S]
    const float* __restrict__ temp,          // scalar
    unsigned short* __restrict__ O) {        // [B,S,E] bf16 token-major
  __shared__ unsigned short Plds[8][16 * 32];   // per-wave P staging (8 KB)

  const int tid  = threadIdx.x;
  const int wave = tid >> 5, lane = tid & 31;
  const int half = lane >> 4, nl = lane & 15;
  const int bh = blockIdx.y;                       // 0..B*H-1
  const int q0 = blockIdx.x * 128 + wave * 16;
  const float invT = 1.0f / temp[0];

  // persistent Q fragments (16x64 split into two K=32 chunks)
  const unsigned short* Qbase = Qh + (bh * SEQ + q0) * HDIM;
  const v16bf qf0 = load_a_frag(Qbase, HDIM);
  const v16bf qf1 = load_a_frag(Qbase + 32, HDIM);

  float qs[8];
#pragma unroll
  for (int r = 0; r < 8; ++r) qs[r] = qsq[bh * SEQ + q0 + half * 8 + r];

  float mi[8], li[8];
#pragma unroll
  for (int r = 0; r < 8; ++r) { mi[r] = -1e30f; li[r] = 0.f; }
  v8f acc[4] = {};

  const unsigned short* Kbh = Kh + bh * SEQ * HDIM;
  const unsigned short* Vbh = Vt + bh * HDIM * SEQ;
  const float* ksrow = ksq + bh * SEQ;

  // preload K fragments for the first tile
  v16bf kf00 = load_bt_frag(Kbh, HDIM);
  v16bf kf01 = load_bt_frag(Kbh + 32, HDIM);
  v16bf kf10 = load_bt_frag(Kbh + 16 * HDIM, HDIM);
  v16bf kf11 = load_bt_frag(Kbh + 16 * HDIM + 32, HDIM);

  for (int j0 = 0; j0 < SEQ; j0 += 32) {
    // V fragments for this tile: issue early, consume after softmax
    v16bf vf[4];
#pragma unroll
    for (int t = 0; t < 4; ++t)
      vf[t] = load_bt_frag(Vbh + (t * 16) * SEQ + j0, SEQ);

    // ---- scores: Q (16x64) @ K^T (64x32) -> two 16x16 tiles ----
    v8f s0 = {}; s0 = wmma_bf16(qf0, kf00, s0); s0 = wmma_bf16(qf1, kf01, s0);
    v8f s1 = {}; s1 = wmma_bf16(qf0, kf10, s1); s1 = wmma_bf16(qf1, kf11, s1);

    // preload next K tile; overlaps softmax VALU below
    if (j0 + 32 < SEQ) {
      const unsigned short* Kn = Kbh + (j0 + 32) * HDIM;
      kf00 = load_bt_frag(Kn, HDIM);
      kf01 = load_bt_frag(Kn + 32, HDIM);
      kf10 = load_bt_frag(Kn + 16 * HDIM, HDIM);
      kf11 = load_bt_frag(Kn + 16 * HDIM + 32, HDIM);
    }
    if (j0 + 64 < SEQ) {   // HBM->L2 one further tile ahead
      __builtin_prefetch(Kbh + (j0 + 64) * HDIM + lane * HDIM, 0, 3);
      __builtin_prefetch(Vbh + nl * SEQ + j0 + 64, 0, 3);
    }

    const float ks0 = ksrow[j0 + nl];
    const float ks1 = ksrow[j0 + 16 + nl];

    float p0[8], p1[8], scale[8];
#pragma unroll
    for (int r = 0; r < 8; ++r) {
      const float d0 = fmaxf(qs[r] + ks0 - 2.f * s0[r], 0.f);
      const float d1 = fmaxf(qs[r] + ks1 - 2.f * s1[r], 0.f);
      const float l0 = -d0 * invT;
      const float l1 = -d1 * invT;
      // row max over the 32 keys (rows live in 16-lane halves)
      float rm = fmaxf(l0, l1);
#pragma unroll
      for (int off = 8; off >= 1; off >>= 1)
        rm = fmaxf(rm, __shfl_xor(rm, off, 32));
      const float mn = fmaxf(mi[r], rm);
      p0[r] = __expf(l0 - mn);
      p1[r] = __expf(l1 - mn);
      float rs = p0[r] + p1[r];
#pragma unroll
      for (int off = 8; off >= 1; off >>= 1)
        rs += __shfl_xor(rs, off, 32);
      scale[r] = __expf(mi[r] - mn);
      li[r] = li[r] * scale[r] + rs;
      mi[r] = mn;
    }
    // rescale running accumulators
#pragma unroll
    for (int t = 0; t < 4; ++t)
#pragma unroll
      for (int r = 0; r < 8; ++r) acc[t][r] *= scale[r];

    // ---- P (16x32) C-layout -> A-layout via per-wave LDS staging ----
    unsigned short* P = Plds[wave];
#pragma unroll
    for (int r = 0; r < 8; ++r) {
      const int m = half * 8 + r;
      P[m * 32 + nl]      = f2bf_bits(p0[r]);
      P[m * 32 + 16 + nl] = f2bf_bits(p1[r]);
    }
    const v16bf pf = load_a_frag(P, 32);   // DS ops are in-order per wave

    // ---- O += P @ V : Vt rows are [d][s] -> contiguous along keys ----
#pragma unroll
    for (int t = 0; t < 4; ++t)
      acc[t] = wmma_bf16(pf, vf[t], acc[t]);
  }

  // epilogue: token-major bf16 [B,S,E], normalize by li (one rcp per row)
  const int b = bh >> 4, h = bh & (HEADS - 1);
#pragma unroll
  for (int r = 0; r < 8; ++r) {
    const float inv = 1.0f / li[r];
    const int s = q0 + half * 8 + r;
#pragma unroll
    for (int t = 0; t < 4; ++t) {
      const float v = acc[t][r] * inv;
      O[(b * SEQ + s) * EMBED + h * HDIM + t * 16 + nl] = f2bf_bits(v);
    }
  }
}

// ------------------------------ launcher -----------------------------------
extern "C" void kernel_launch(void* const* d_in, const int* in_sizes, int n_in,
                              void* d_out, int out_size, void* d_ws, size_t ws_size,
                              hipStream_t stream) {
  const float* query = (const float*)d_in[0];
  const float* wq = (const float*)d_in[1];
  const float* bq = (const float*)d_in[2];
  const float* wk = (const float*)d_in[3];
  const float* bk = (const float*)d_in[4];
  const float* wv = (const float*)d_in[5];
  const float* bv = (const float*)d_in[6];
  const float* wo = (const float*)d_in[7];
  const float* bo = (const float*)d_in[8];
  const float* temp = (const float*)d_in[9];
  (void)in_sizes; (void)n_in; (void)out_size; (void)ws_size;

  char* ws = (char*)d_ws;
  size_t off = 0;
  auto carve = [&](size_t bytes) -> void* {
    void* p = ws + off;
    off += (bytes + 255) & ~(size_t)255;
    return p;
  };
  const size_t actBytes  = (size_t)MTOT * EMBED * 2;              // 8 MB
  const size_t wBytes    = (size_t)EMBED * EMBED * 2;             // 2 MB
  const size_t headBytes = (size_t)BATCH * HEADS * SEQ * HDIM * 2;// 8 MB
  const size_t sqBytes   = (size_t)BATCH * HEADS * SEQ * 4;       // 256 KB

  unsigned short* qbf  = (unsigned short*)carve(actBytes);
  unsigned short* wqt  = (unsigned short*)carve(wBytes);   // [N,K] bf16
  unsigned short* wkt  = (unsigned short*)carve(wBytes);
  unsigned short* wvt  = (unsigned short*)carve(wBytes);
  unsigned short* wot  = (unsigned short*)carve(wBytes);
  unsigned short* Qh   = (unsigned short*)carve(headBytes);
  unsigned short* Kh   = (unsigned short*)carve(headBytes);
  unsigned short* Vtb  = (unsigned short*)carve(headBytes); // [B,H,D,S]
  float*          qsq  = (float*)carve(sqBytes);
  float*          ksq  = (float*)carve(sqBytes);
  unsigned short* attO = (unsigned short*)carve(actBytes);

  const int nAct = MTOT * EMBED;        // 4,194,304

  // 1) fp32 -> bf16 activations; weights transposed+converted to [N,K]
  cvt_bf16_kernel<<<(nAct + 255) / 256, 256, 0, stream>>>(query, qbf, nAct);
  dim3 tgrid(EMBED / 32, EMBED / 32);   // (32, 32)
  transpose_cvt_kernel<<<tgrid, 256, 0, stream>>>(wq, wqt);
  transpose_cvt_kernel<<<tgrid, 256, 0, stream>>>(wk, wkt);
  transpose_cvt_kernel<<<tgrid, 256, 0, stream>>>(wv, wvt);
  transpose_cvt_kernel<<<tgrid, 256, 0, stream>>>(wo, wot);

  // 2) Q/K/V projections
  dim3 ggrid(EMBED / 64, MTOT / 128);   // (16, 32)
  gemm_bf16_kernel<<<ggrid, 256, 0, stream>>>(qbf, wqt, bq, nullptr, Qh, 1);
  gemm_bf16_kernel<<<ggrid, 256, 0, stream>>>(qbf, wkt, bk, nullptr, Kh, 1);
  gemm_bf16_kernel<<<ggrid, 256, 0, stream>>>(qbf, wvt, bv, nullptr, Vtb, 2);

  // 3) squared norms of Q and K rows
  const int nRows = BATCH * HEADS * SEQ;  // 65,536
  rowsq_kernel<<<nRows / 256, 256, 0, stream>>>(Qh, qsq);
  rowsq_kernel<<<nRows / 256, 256, 0, stream>>>(Kh, ksq);

  // 4) Euclidean flash attention
  dim3 agrid(SEQ / 128, BATCH * HEADS);   // (16, 32)
  attn_kernel<<<agrid, 256, 0, stream>>>(Qh, Kh, Vtb, qsq, ksq, temp, attO);

  // 5) output projection -> fp32 d_out
  gemm_bf16_kernel<<<ggrid, 256, 0, stream>>>(attO, wot, bo, (float*)d_out,
                                              nullptr, 0);
}
